// Decoder_57664230916439
// MI455X (gfx1250) — matrix operand order repaired and checked by reading, not compiled, not run on hardware
//
#include <hip/hip_runtime.h>

// ---------------------------------------------------------------------------
// Types for CDNA5 WMMA (wave32)
// ---------------------------------------------------------------------------
typedef __attribute__((ext_vector_type(16))) __bf16          bf16x16;
typedef __attribute__((ext_vector_type(8)))  float           v8f;
typedef __attribute__((ext_vector_type(8)))  unsigned short  u16x8;
typedef __attribute__((ext_vector_type(16))) unsigned short  u16x16;

#define B_      32
#define T_IN_   256
#define T_OUT_  768
#define N_MEL_  80
#define ENC_    512
#define RNN_    1024
#define GATES_N 4096
#define PRE_    256
#define ATTD_   128
#define LOCF_   32
#define LOCK_   31
#define NWG_    64          // persistent decoder grid
#define NTHR_   (NWG_ * 256)
#define KCH_    384         // K-chunk staged into LDS per double-buffer slot

__device__ inline unsigned short f32_to_bf16(float x) {
    unsigned int u = __float_as_uint(x);
    unsigned int r = (u + 0x7FFFu + ((u >> 16) & 1u)) >> 16;
    return (unsigned short)r;
}
__device__ inline float sigm(float x) { return 1.f / (1.f + __expf(-x)); }

__device__ inline v8f vzero8() {
    v8f z;
#pragma unroll
    for (int i = 0; i < 8; ++i) z[i] = 0.f;
    return z;
}

// A-matrix 16x32 bf16 fragment (wave32, contiguous k order matching load_b_frag):
// lanes 0-15:  row = mbase+lane,    K = kb+[0..15]
// lanes 16-31: row = mbase+lane-16, K = kb+[16..31]
__device__ inline bf16x16 load_a_frag(const unsigned short* A, int ldk,
                                      int mbase, int kbase) {
    int lane = threadIdx.x & 31;
    int m  = mbase + (lane & 15);
    int k0 = kbase + ((lane & 16) ? 16 : 0);
    const unsigned short* p = A + (size_t)m * ldk + k0;
    u16x8 lo = *reinterpret_cast<const u16x8*>(p);
    u16x8 hi = *reinterpret_cast<const u16x8*>(p + 8);
    u16x16 raw;
#pragma unroll
    for (int i = 0; i < 8; ++i) { raw[i] = lo[i]; raw[8 + i] = hi[i]; }
    return __builtin_bit_cast(bf16x16, raw);
}

// B-matrix 32x16 bf16 fragment. Weight stored row-major [N][K] (reference
// layout, no transpose): lane n gets a contiguous K run -> one 32B load.
__device__ inline bf16x16 load_b_frag(const unsigned short* W, int ldk,
                                      int nbase, int kbase) {
    int lane = threadIdx.x & 31;
    int n = nbase + (lane & 15);
    int k = kbase + ((lane & 16) ? 16 : 0);
    u16x16 raw = *reinterpret_cast<const u16x16*>(W + (size_t)n * ldk + k);
    return __builtin_bit_cast(bf16x16, raw);
}

__device__ inline v8f wmma_bf16(bf16x16 a, bf16x16 b, v8f c) {
    return __builtin_amdgcn_wmma_f32_16x16x32_bf16(
        false, a, false, b, (short)0, c, false, false);
}

// ---------------------------------------------------------------------------
// Async global->LDS staging (gfx1250 GLOBAL_LOAD_ASYNC_TO_LDS_B128, ASYNCcnt)
// ---------------------------------------------------------------------------
__device__ inline void async_b128_to_lds(unsigned short* lds_dst,
                                         const unsigned short* gsrc) {
    unsigned lds_off = (unsigned)(size_t)(void*)lds_dst;  // low 32b = LDS addr
    asm volatile("global_load_async_to_lds_b128 %0, %1, off"
                 :: "v"(lds_off), "v"(gsrc) : "memory");
}
__device__ inline void wait_async0() {
    asm volatile("s_wait_asynccnt 0x0" ::: "memory");
}

// Stage 32 rows x clen cols (bf16) of A (row stride ldk) into LDS slot
// (row stride KCH_), cooperatively across the 256-thread block.
__device__ inline void stage_chunk(unsigned short* sdst,
                                   const unsigned short* A, int ldk,
                                   int kb, int clen) {
    const int vpr  = clen / 8;       // 16B vectors per row
    const int nvec = 32 * vpr;
    for (int c = threadIdx.x; c < nvec; c += 256) {
        int row = c / vpr, kv = c - row * vpr;
        async_b128_to_lds(sdst + row * KCH_ + kv * 8,
                          A + (size_t)row * ldk + kb + kv * 8);
    }
}

// ---------------------------------------------------------------------------
// Weight conversion kernels
// ---------------------------------------------------------------------------
__global__ void k_f32_to_bf16(const float* __restrict__ src,
                              unsigned short* __restrict__ dst, int n) {
    int i = blockIdx.x * blockDim.x + threadIdx.x;
    if (i < n) dst[i] = f32_to_bf16(src[i]);
}

// prenet_w1 (256,80) -> bf16 (256,96), zero-padded K
__global__ void k_w1_pad(const float* __restrict__ src,
                         unsigned short* __restrict__ dst) {
    int i = blockIdx.x * blockDim.x + threadIdx.x;
    if (i < 256 * 96) {
        int n = i / 96, k = i % 96;
        dst[i] = (k < 80) ? f32_to_bf16(src[n * 80 + k]) : (unsigned short)0;
    }
}

// proj_w (80,1536) + gate_w (1,1536) -> bf16 [96][1536], rows 81..95 zero
__global__ void k_pg_pad(const float* __restrict__ pw,
                         const float* __restrict__ gw,
                         unsigned short* __restrict__ dst) {
    int i = blockIdx.x * blockDim.x + threadIdx.x;
    if (i < 96 * 1536) {
        int n = i / 1536, k = i % 1536;
        float v = 0.f;
        if (n < 80) v = pw[n * 1536 + k];
        else if (n == 80) v = gw[k];
        dst[i] = f32_to_bf16(v);
    }
}

// ---------------------------------------------------------------------------
// Prenet for all timesteps: xs_b[t][b][256], WMMA, one WG per timestep
// ---------------------------------------------------------------------------
__global__ __launch_bounds__(256) void k_prenet(
    const float* __restrict__ din,            // [32][80][768]
    const unsigned short* __restrict__ w1b,   // [256][96]
    const unsigned short* __restrict__ w2b,   // [256][256]
    unsigned short* __restrict__ xs_b) {      // [768][32][256]
    __shared__ __align__(32) unsigned short sA[32 * 96];
    __shared__ __align__(32) unsigned short sH[32 * 256];
    const int t   = blockIdx.x;
    const int tid = threadIdx.x;

    for (int i = tid; i < 32 * 96; i += 256) {
        int b = i / 96, k = i % 96;
        float v = 0.f;
        if (t > 0 && k < 80) v = din[((size_t)b * 80 + k) * 768 + (t - 1)];
        sA[i] = f32_to_bf16(v);
    }
    __syncthreads();

    const int w    = tid >> 5;
    const int lane = tid & 31;

#pragma unroll
    for (int mt = 0; mt < 2; ++mt)
#pragma unroll
        for (int rep = 0; rep < 2; ++rep) {
            int mbase = mt * 16, nbase = w * 32 + rep * 16;
            v8f acc = vzero8();
#pragma unroll
            for (int kb = 0; kb < 96; kb += 32)
                acc = wmma_bf16(load_a_frag(sA, 96, mbase, kb),
                                load_b_frag(w1b, 96, nbase, kb), acc);
            int n = nbase + (lane & 15);
#pragma unroll
            for (int r = 0; r < 8; ++r) {
                int m = mbase + r + ((lane & 16) ? 8 : 0);
                sH[m * 256 + n] = f32_to_bf16(fmaxf(acc[r], 0.f));
            }
        }
    __syncthreads();

#pragma unroll
    for (int mt = 0; mt < 2; ++mt)
#pragma unroll
        for (int rep = 0; rep < 2; ++rep) {
            int mbase = mt * 16, nbase = w * 32 + rep * 16;
            v8f acc = vzero8();
#pragma unroll
            for (int kb = 0; kb < 256; kb += 32)
                acc = wmma_bf16(load_a_frag(sH, 256, mbase, kb),
                                load_b_frag(w2b, 256, nbase, kb), acc);
            int n = nbase + (lane & 15);
#pragma unroll
            for (int r = 0; r < 8; ++r) {
                int m = mbase + r + ((lane & 16) ? 8 : 0);
                xs_b[((size_t)t * 32 + m) * 256 + n] =
                    f32_to_bf16(fmaxf(acc[r], 0.f));
            }
        }
}

// ---------------------------------------------------------------------------
// processed_memory = memory @ memory_w.T  (8192x512 x 512->128), WMMA
// ---------------------------------------------------------------------------
__global__ __launch_bounds__(256) void k_pmem(
    const unsigned short* __restrict__ mem_b,   // [8192][512] bf16
    const unsigned short* __restrict__ memw_b,  // [128][512] bf16
    float* __restrict__ pmem) {                 // [8192][128] f32
    const int g    = blockIdx.x;
    const int w    = threadIdx.x >> 5;
    const int lane = threadIdx.x & 31;
    const unsigned short* A = mem_b + (size_t)g * 32 * 512;
#pragma unroll
    for (int rep = 0; rep < 2; ++rep) {
        int mbase = (w & 1) * 16;
        int nbase = ((w >> 1) + rep * 4) * 16;
        v8f acc = vzero8();
        for (int kb = 0; kb < 512; kb += 32)
            acc = wmma_bf16(load_a_frag(A, 512, mbase, kb),
                            load_b_frag(memw_b, 512, nbase, kb), acc);
        int n = nbase + (lane & 15);
#pragma unroll
        for (int r = 0; r < 8; ++r) {
            int m = mbase + r + ((lane & 16) ? 8 : 0);
            pmem[((size_t)g * 32 + m) * 128 + n] = acc[r];
        }
    }
}

// ---------------------------------------------------------------------------
// Persistent decoder
// ---------------------------------------------------------------------------
struct DecParams {
    const float *memory, *bih_a, *bhh_a, *bih_d, *bhh_d;
    const float *loc_conv_w, *loc_dense_w, *v_w, *proj_b, *gate_b;
    const unsigned short *att_wih_b, *att_whh_b, *dec_wih_b, *dec_whh_b;
    const unsigned short *xs_b, *qw_b, *pg_b;
    const float *pmem;
    float *gates, *hatt, *catt, *hdec, *cdec, *ctx, *pq, *aw, *aw_cum, *energ;
    unsigned short *hatt_b, *hdec_b, *xcat, *xcat2;
    float *mel_out, *gate_out, *align_out;
    unsigned *bar;
};

__device__ inline void grid_barrier(unsigned* bar, unsigned& gen) {
    __syncthreads();
    if (threadIdx.x == 0) {
        __threadfence();
        atomicAdd(bar, 1u);
        const unsigned target = (gen + 1u) * (unsigned)NWG_;
        while (__atomic_load_n(bar, __ATOMIC_ACQUIRE) < target)
            __builtin_amdgcn_s_sleep(2);
    }
    ++gen;
    __syncthreads();
}

// gates[32][4096] = A1[32][K1] @ W1[4096][K1]^T + A2[32][K2] @ W2[4096][K2]^T
// A is staged into double-buffered LDS via async-to-LDS copies (overlapped
// with WMMA on the previous chunk); B streams from L2-resident bf16 weights.
__device__ inline void gemm_gates(const unsigned short* A1, int K1,
                                  const unsigned short* W1,
                                  const unsigned short* A2, int K2,
                                  const unsigned short* W2,
                                  float* gates, unsigned short* sA) {
    const int w     = threadIdx.x >> 5;
    const int lane  = threadIdx.x & 31;
    const int mbase = (w & 1) * 16;
    const int nbase = blockIdx.x * 64 + (w >> 1) * 16;
    const int nch1  = (K1 + KCH_ - 1) / KCH_;
    const int nch   = nch1 + (K2 + KCH_ - 1) / KCH_;

    v8f acc = vzero8();
    {   // prologue: stage chunk 0
        int clen0 = (K1 < KCH_) ? K1 : KCH_;
        stage_chunk(sA, A1, K1, 0, clen0);
    }
    int buf = 0;
    for (int ci = 0; ci < nch; ++ci) {
        const unsigned short* A; const unsigned short* W; int K, kb0;
        if (ci < nch1) { A = A1; W = W1; K = K1; kb0 = ci * KCH_; }
        else           { A = A2; W = W2; K = K2; kb0 = (ci - nch1) * KCH_; }
        int clen = K - kb0; if (clen > KCH_) clen = KCH_;

        wait_async0();
        __syncthreads();                 // chunk ci resident in sA[buf]
        if (ci + 1 < nch) {              // overlap: stage chunk ci+1
            const unsigned short* An; int Kn, kbn;
            if (ci + 1 < nch1) { An = A1; Kn = K1; kbn = (ci + 1) * KCH_; }
            else { An = A2; Kn = K2; kbn = (ci + 1 - nch1) * KCH_; }
            int clenn = Kn - kbn; if (clenn > KCH_) clenn = KCH_;
            stage_chunk(sA + (buf ^ 1) * (32 * KCH_), An, Kn, kbn, clenn);
        }
        const unsigned short* sbuf = sA + buf * (32 * KCH_);
        for (int kk = 0; kk < clen; kk += 32) {
            if (kb0 + kk + 64 <= K)
                __builtin_prefetch(
                    W + (size_t)(nbase + lane) * K + kb0 + kk + 32, 0, 1);
            acc = wmma_bf16(load_a_frag(sbuf, KCH_, mbase, kk),
                            load_b_frag(W, K, nbase, kb0 + kk), acc);
        }
        __syncthreads();                 // done reading sA[buf]
        buf ^= 1;
    }
    int n = nbase + (lane & 15);
#pragma unroll
    for (int r = 0; r < 8; ++r) {
        int m = mbase + r + ((lane & 16) ? 8 : 0);
        gates[m * GATES_N + n] = acc[r];
    }
}

__global__ __launch_bounds__(256) void k_decoder(DecParams p) {
    __shared__ __align__(32) unsigned short sA[2 * 32 * KCH_];  // 48 KB
    __shared__ float sred[256];
    __shared__ float s_aw[256];
    unsigned gen  = 0;
    const int gid = blockIdx.x * 256 + threadIdx.x;

    for (int t = 0; t < T_OUT_; ++t) {
        // ---- phase 0: xcat[:, :256] = xs[t] -----------------------------
        if (gid < B_ * PRE_) {
            int b = gid >> 8, k = gid & 255;
            p.xcat[b * 768 + k] = p.xs_b[((size_t)t * B_ + b) * 256 + k];
        }
        grid_barrier(p.bar, gen);

        // ---- phase 1: attention-LSTM gates GEMM (async-staged A) --------
        gemm_gates(p.xcat, 768, p.att_wih_b, p.hatt_b, 1024, p.att_whh_b,
                   p.gates, sA);
        grid_barrier(p.bar, gen);

        // ---- phase 2: attention-LSTM elementwise update -----------------
        for (int idx = gid; idx < B_ * RNN_; idx += NTHR_) {
            int b = idx >> 10, j = idx & 1023;
            const float* g0 = p.gates + b * GATES_N;
            float ig = g0[j]        + p.bih_a[j]        + p.bhh_a[j];
            float fg = g0[1024 + j] + p.bih_a[1024 + j] + p.bhh_a[1024 + j];
            float gg = g0[2048 + j] + p.bih_a[2048 + j] + p.bhh_a[2048 + j];
            float og = g0[3072 + j] + p.bih_a[3072 + j] + p.bhh_a[3072 + j];
            float c  = sigm(fg) * p.catt[idx] + sigm(ig) * tanhf(gg);
            float h  = sigm(og) * tanhf(c);
            p.catt[idx] = c;
            p.hatt[idx] = h;
            unsigned short hb = f32_to_bf16(h);
            p.hatt_b[idx] = hb;
            p.xcat2[b * 1536 + j] = hb;
        }
        grid_barrier(p.bar, gen);

        // ---- phase 2b: pq = att_h @ query_w.T (WMMA, WGs 0-1) -----------
        if (blockIdx.x < 2) {
            const int w = threadIdx.x >> 5, lane = threadIdx.x & 31;
            int mbase = (w & 1) * 16;
            int nbase = blockIdx.x * 64 + (w >> 1) * 16;
            v8f acc = vzero8();
            for (int kb = 0; kb < RNN_; kb += 32)
                acc = wmma_bf16(load_a_frag(p.hatt_b, RNN_, mbase, kb),
                                load_b_frag(p.qw_b, RNN_, nbase, kb), acc);
            int n = nbase + (lane & 15);
#pragma unroll
            for (int r = 0; r < 8; ++r) {
                int b = mbase + r + ((lane & 16) ? 8 : 0);
                p.pq[b * ATTD_ + n] = acc[r];
            }
        }
        grid_barrier(p.bar, gen);

        // ---- phase 3: location conv + attention energies ----------------
        if (gid < B_ * T_IN_) {
            int b = gid >> 8, tt = gid & 255;
            float locf[LOCF_];
#pragma unroll 4
            for (int f = 0; f < LOCF_; ++f) {
                const float* wf = p.loc_conv_w + f * 62;
                float s = 0.f;
                for (int k = 0; k < LOCK_; ++k) {
                    int ti = tt + k - 15;
                    if (ti >= 0 && ti < T_IN_) {
                        s += wf[k]         * p.aw[b * T_IN_ + ti];
                        s += wf[LOCK_ + k] * p.aw_cum[b * T_IN_ + ti];
                    }
                }
                locf[f] = s;
            }
            const float* pm  = p.pmem + ((size_t)b * T_IN_ + tt) * ATTD_;
            const float* pqb = p.pq + b * ATTD_;
            float e = 0.f;
            for (int d = 0; d < ATTD_; ++d) {
                const float* ld = p.loc_dense_w + d * LOCF_;
                float a = pqb[d] + pm[d];
#pragma unroll
                for (int f = 0; f < LOCF_; ++f) a += ld[f] * locf[f];
                e += p.v_w[d] * tanhf(a);
            }
            p.energ[b * T_IN_ + tt] = e;
        }
        grid_barrier(p.bar, gen);

        // ---- phase 4: softmax + context + cumulative attention ----------
        if (blockIdx.x < B_) {
            int b = blockIdx.x, tt = threadIdx.x;
            float e = p.energ[b * T_IN_ + tt];
            sred[tt] = e;
            __syncthreads();
            for (int s2 = 128; s2 > 0; s2 >>= 1) {
                if (tt < s2) sred[tt] = fmaxf(sred[tt], sred[tt + s2]);
                __syncthreads();
            }
            float mx = sred[0];
            __syncthreads();
            float pe = __expf(e - mx);
            sred[tt] = pe;
            __syncthreads();
            for (int s2 = 128; s2 > 0; s2 >>= 1) {
                if (tt < s2) sred[tt] += sred[tt + s2];
                __syncthreads();
            }
            float awv = pe / sred[0];
            s_aw[tt] = awv;
            p.aw[b * T_IN_ + tt] = awv;
            p.aw_cum[b * T_IN_ + tt] += awv;
            p.align_out[((size_t)b * T_OUT_ + t) * T_IN_ + tt] = awv;
            __syncthreads();
            for (int e2 = tt; e2 < ENC_; e2 += 256) {
                const float* mb = p.memory + (size_t)b * T_IN_ * ENC_ + e2;
                float s = 0.f;
                for (int q = 0; q < T_IN_; ++q) s += s_aw[q] * mb[q * ENC_];
                p.ctx[b * ENC_ + e2] = s;
                unsigned short cb = f32_to_bf16(s);
                p.xcat[b * 768 + 256 + e2]    = cb;
                p.xcat2[b * 1536 + 1024 + e2] = cb;
            }
        }
        grid_barrier(p.bar, gen);

        // ---- phase 5: decoder-LSTM gates GEMM (async-staged A) ----------
        gemm_gates(p.xcat2, 1536, p.dec_wih_b, p.hdec_b, 1024, p.dec_whh_b,
                   p.gates, sA);
        grid_barrier(p.bar, gen);

        // ---- phase 6: decoder-LSTM elementwise update -------------------
        for (int idx = gid; idx < B_ * RNN_; idx += NTHR_) {
            int b = idx >> 10, j = idx & 1023;
            const float* g0 = p.gates + b * GATES_N;
            float ig = g0[j]        + p.bih_d[j]        + p.bhh_d[j];
            float fg = g0[1024 + j] + p.bih_d[1024 + j] + p.bhh_d[1024 + j];
            float gg = g0[2048 + j] + p.bih_d[2048 + j] + p.bhh_d[2048 + j];
            float og = g0[3072 + j] + p.bih_d[3072 + j] + p.bhh_d[3072 + j];
            float c  = sigm(fg) * p.cdec[idx] + sigm(ig) * tanhf(gg);
            float h  = sigm(og) * tanhf(c);
            p.cdec[idx]  = c;
            p.hdec[idx]  = h;
            p.hdec_b[idx] = f32_to_bf16(h);
        }
        grid_barrier(p.bar, gen);

        // ---- phase 7: projection + gate (WMMA [32x1536]x[1536->96]) -----
        if (blockIdx.x < 3) {
            const int w = threadIdx.x >> 5;
            if (w < 4) {
                const int lane = threadIdx.x & 31;
                int mbase = (w & 1) * 16;
                int nbase = blockIdx.x * 32 + (w >> 1) * 16;
                v8f acc = vzero8();
                // dha = [hdec | ctx]: hdec_b [32][1024]; ctx bf16 already
                // lives at xcat2[:,1024:1536] (row stride 1536)
                for (int kb = 0; kb < RNN_; kb += 32)
                    acc = wmma_bf16(load_a_frag(p.hdec_b, RNN_, mbase, kb),
                                    load_b_frag(p.pg_b, 1536, nbase, kb), acc);
                for (int kb = 0; kb < ENC_; kb += 32)
                    acc = wmma_bf16(
                        load_a_frag(p.xcat2 + 1024, 1536, mbase, kb),
                        load_b_frag(p.pg_b, 1536, nbase, RNN_ + kb), acc);
                int n = nbase + (lane & 15);
#pragma unroll
                for (int r = 0; r < 8; ++r) {
                    int b = mbase + r + ((lane & 16) ? 8 : 0);
                    float v = acc[r];
                    if (n < N_MEL_)
                        p.mel_out[((size_t)b * N_MEL_ + n) * T_OUT_ + t] =
                            v + p.proj_b[n];
                    else if (n == N_MEL_)
                        p.gate_out[b * T_OUT_ + t] = v + p.gate_b[0];
                }
            }
        }
        grid_barrier(p.bar, gen);
    }
}

// ---------------------------------------------------------------------------
// Host-side orchestration
// ---------------------------------------------------------------------------
extern "C" void kernel_launch(void* const* d_in, const int* in_sizes, int n_in,
                              void* d_out, int out_size, void* d_ws,
                              size_t ws_size, hipStream_t stream) {
    const float* memory = (const float*)d_in[0];
    const float* din    = (const float*)d_in[1];
    const float* w1     = (const float*)d_in[2];
    const float* w2     = (const float*)d_in[3];
    const float* a_wih  = (const float*)d_in[4];
    const float* a_whh  = (const float*)d_in[5];
    const float* a_bih  = (const float*)d_in[6];
    const float* a_bhh  = (const float*)d_in[7];
    const float* qw     = (const float*)d_in[8];
    const float* mw     = (const float*)d_in[9];
    const float* vw     = (const float*)d_in[10];
    const float* lcw    = (const float*)d_in[11];
    const float* ldw    = (const float*)d_in[12];
    const float* d_wih  = (const float*)d_in[13];
    const float* d_whh  = (const float*)d_in[14];
    const float* d_bih  = (const float*)d_in[15];
    const float* d_bhh  = (const float*)d_in[16];
    const float* pw     = (const float*)d_in[17];
    const float* pb     = (const float*)d_in[18];
    const float* gw     = (const float*)d_in[19];
    const float* gb     = (const float*)d_in[20];

    char* ws = (char*)d_ws;
    size_t off = 0;
    auto take = [&](size_t bytes) -> size_t {
        size_t o = off;
        off += (bytes + 255) & ~(size_t)255;
        return o;
    };
    size_t o_awih = take((size_t)4096 * 768 * 2);
    size_t o_awhh = take((size_t)4096 * 1024 * 2);
    size_t o_dwih = take((size_t)4096 * 1536 * 2);
    size_t o_dwhh = take((size_t)4096 * 1024 * 2);
    size_t o_w1b  = take((size_t)256 * 96 * 2);
    size_t o_w2b  = take((size_t)256 * 256 * 2);
    size_t o_mwb  = take((size_t)128 * 512 * 2);
    size_t o_qwb  = take((size_t)128 * 1024 * 2);
    size_t o_pgb  = take((size_t)96 * 1536 * 2);
    size_t o_memb = take((size_t)32 * 256 * 512 * 2);
    size_t o_xsb  = take((size_t)768 * 32 * 256 * 2);
    size_t o_pmem = take((size_t)32 * 256 * 128 * 4);
    size_t o_gate = take((size_t)32 * 4096 * 4);
    // zero-initialized recurrent state block (contiguous -> one memset)
    size_t o_state = off;
    size_t o_hatt  = take((size_t)32 * 1024 * 4);
    size_t o_catt  = take((size_t)32 * 1024 * 4);
    size_t o_hdec  = take((size_t)32 * 1024 * 4);
    size_t o_cdec  = take((size_t)32 * 1024 * 4);
    size_t o_hattb = take((size_t)32 * 1024 * 2);
    size_t o_hdecb = take((size_t)32 * 1024 * 2);
    size_t o_xcat  = take((size_t)32 * 768 * 2);
    size_t o_xcat2 = take((size_t)32 * 1536 * 2);
    size_t o_ctx   = take((size_t)32 * 512 * 4);
    size_t o_pq    = take((size_t)32 * 128 * 4);
    size_t o_aw    = take((size_t)32 * 256 * 4);
    size_t o_awcum = take((size_t)32 * 256 * 4);
    size_t o_energ = take((size_t)32 * 256 * 4);
    size_t o_bar   = take(256);
    size_t state_bytes = off - o_state;
    (void)ws_size; (void)n_in; (void)in_sizes; (void)out_size;

    hipMemsetAsync(ws + o_state, 0, state_bytes, stream);

    auto cvt = [&](const float* src, size_t o_dst, size_t n) {
        k_f32_to_bf16<<<(unsigned)((n + 255) / 256), 256, 0, stream>>>(
            src, (unsigned short*)(ws + o_dst), (int)n);
    };
    cvt(a_wih,  o_awih, (size_t)4096 * 768);
    cvt(a_whh,  o_awhh, (size_t)4096 * 1024);
    cvt(d_wih,  o_dwih, (size_t)4096 * 1536);
    cvt(d_whh,  o_dwhh, (size_t)4096 * 1024);
    cvt(w2,     o_w2b,  (size_t)256 * 256);
    cvt(mw,     o_mwb,  (size_t)128 * 512);
    cvt(qw,     o_qwb,  (size_t)128 * 1024);
    cvt(memory, o_memb, (size_t)32 * 256 * 512);
    k_w1_pad<<<(256 * 96 + 255) / 256, 256, 0, stream>>>(
        w1, (unsigned short*)(ws + o_w1b));
    k_pg_pad<<<(96 * 1536 + 255) / 256, 256, 0, stream>>>(
        pw, gw, (unsigned short*)(ws + o_pgb));

    k_prenet<<<T_OUT_, 256, 0, stream>>>(
        din, (const unsigned short*)(ws + o_w1b),
        (const unsigned short*)(ws + o_w2b), (unsigned short*)(ws + o_xsb));

    k_pmem<<<256, 256, 0, stream>>>(
        (const unsigned short*)(ws + o_memb),
        (const unsigned short*)(ws + o_mwb), (float*)(ws + o_pmem));

    float* out = (float*)d_out;
    DecParams P;
    P.memory = memory;
    P.bih_a = a_bih;  P.bhh_a = a_bhh;
    P.bih_d = d_bih;  P.bhh_d = d_bhh;
    P.loc_conv_w = lcw; P.loc_dense_w = ldw; P.v_w = vw;
    P.proj_b = pb;      P.gate_b = gb;
    P.att_wih_b = (const unsigned short*)(ws + o_awih);
    P.att_whh_b = (const unsigned short*)(ws + o_awhh);
    P.dec_wih_b = (const unsigned short*)(ws + o_dwih);
    P.dec_whh_b = (const unsigned short*)(ws + o_dwhh);
    P.xs_b = (const unsigned short*)(ws + o_xsb);
    P.qw_b = (const unsigned short*)(ws + o_qwb);
    P.pg_b = (const unsigned short*)(ws + o_pgb);
    P.pmem  = (const float*)(ws + o_pmem);
    P.gates = (float*)(ws + o_gate);
    P.hatt  = (float*)(ws + o_hatt);  P.catt = (float*)(ws + o_catt);
    P.hdec  = (float*)(ws + o_hdec);  P.cdec = (float*)(ws + o_cdec);
    P.ctx   = (float*)(ws + o_ctx);   P.pq   = (float*)(ws + o_pq);
    P.aw    = (float*)(ws + o_aw);    P.aw_cum = (float*)(ws + o_awcum);
    P.energ = (float*)(ws + o_energ);
    P.hatt_b = (unsigned short*)(ws + o_hattb);
    P.hdec_b = (unsigned short*)(ws + o_hdecb);
    P.xcat   = (unsigned short*)(ws + o_xcat);
    P.xcat2  = (unsigned short*)(ws + o_xcat2);
    P.mel_out   = out;
    P.gate_out  = out + (size_t)B_ * N_MEL_ * T_OUT_;
    P.align_out = out + (size_t)B_ * N_MEL_ * T_OUT_ + (size_t)B_ * T_OUT_;
    P.bar = (unsigned*)(ws + o_bar);

    k_decoder<<<NWG_, 256, 0, stream>>>(P);
}
